// MazeGNN_1305670058224
// MI455X (gfx1250) — compile-verified, matrix operand order checked by reading, not run on hardware
//
#include <hip/hip_runtime.h>
#include <hip/hip_bf16.h>

#define N_NODES 2500
#define NP      2512      // padded to multiple of 16 (157 tiles)
#define N_EDGES 10000     // 625 tiles of 16
#define HID     64
#define DD      128
#define BN_EPS  1e-5f
#define ITERS   150       // ceil(3*sqrt(2500))

typedef __attribute__((ext_vector_type(16))) _Float16 v16h;
typedef __attribute__((ext_vector_type(8)))  float    v8f;

// ---------------------------------------------------------------- utilities
__global__ void k_zero_f32(float* p, int n) {
  int i = blockIdx.x * blockDim.x + threadIdx.x;
  if (i < n) p[i] = 0.f;
}

// convert f32 -> f16 and re-zero the source (re-arms Agg for next scatter)
__global__ void k_f32_to_f16_zero(float* __restrict__ in, _Float16* __restrict__ out, int n) {
  int i = blockIdx.x * blockDim.x + threadIdx.x;
  if (i < n) { out[i] = (_Float16)in[i]; in[i] = 0.f; }
}

// Bt[n*din + k] = (f16) W[k*dout + n]   (weights constant per launch)
__global__ void k_transpose_f16(const float* __restrict__ W, _Float16* __restrict__ Bt,
                                int din, int dout) {
  int idx = blockIdx.x * blockDim.x + threadIdx.x;
  if (idx >= din * dout) return;
  int n = idx / din, k = idx - n * din;
  Bt[idx] = (_Float16)W[k * dout + n];
}

// ---------------------------------------------------------------- encoder L1 (2 -> 64)
__global__ void k_enc_l1(const float* __restrict__ x, const float* __restrict__ W,
                         const float* __restrict__ b, float* __restrict__ out) {
  int r = blockIdx.x;        // 0..NP-1
  int j = threadIdx.x;       // 0..63
  float v = 0.f;
  if (r < N_NODES) {
    v = x[r * 2 + 0] * W[j] + x[r * 2 + 1] * W[64 + j] + b[j];
    v = fmaxf(v, 0.f);
  }
  out[r * 64 + j] = v;
}

// ---------------------------------------------------------------- BatchNorm pieces
// stats layout (floats): [0..127]=sum  [128..255]=sumsq  [256..383]=scale  [384..511]=shift
__global__ void k_col_reduce(const float* __restrict__ in, int ld, int rows,
                             float* __restrict__ stats, int chunk) {
  int c  = threadIdx.x;                 // blockDim == cols
  int r0 = blockIdx.x * chunk;
  int r1 = min(rows, r0 + chunk);
  float s = 0.f, s2 = 0.f;
  for (int r = r0; r < r1; ++r) { float v = in[r * ld + c]; s += v; s2 += v * v; }
  atomicAdd(&stats[c], s);
  atomicAdd(&stats[128 + c], s2);
}

__global__ void k_bn_finalize(float* __restrict__ stats, const float* __restrict__ g,
                              const float* __restrict__ be, float rows) {
  int c = threadIdx.x;                  // blockDim == cols
  float mean = stats[c] / rows;
  float var  = stats[128 + c] / rows - mean * mean;
  float s    = g[c] * rsqrtf(var + BN_EPS);
  stats[256 + c] = s;
  stats[384 + c] = be[c] - mean * s;
  stats[c] = 0.f; stats[128 + c] = 0.f;  // re-arm accumulators for next layer
}

template <int COLS>
__global__ void k_bn_apply_f16(const float* __restrict__ in, int ldin,
                               _Float16* __restrict__ out, int ldout,
                               const float* __restrict__ stats, int rowsPad) {
  int i = blockIdx.x * blockDim.x + threadIdx.x;
  if (i >= rowsPad * COLS) return;
  int r = i / COLS, c = i - r * COLS;
  out[r * ldout + c] = (_Float16)(in[r * ldin + c] * stats[256 + c] + stats[384 + c]);
}

// ---------------------------------------------------------------- edge gather: X[e]=cat(h[dst],h[src])
__global__ void k_gather(const _Float16* __restrict__ H16, const long long* __restrict__ ei,
                         _Float16* __restrict__ X) {
  int t = blockIdx.x * blockDim.x + threadIdx.x;    // E*32 threads
  int e = t >> 5, q = t & 31;
  int node = (int)((q < 16) ? ei[N_EDGES + e]       // x_i = h[dst]
                            : ei[e]);               // x_j = h[src]
  const uint4* Hv = (const uint4*)H16;              // 16 x uint4 per 128-f16 row
  uint4*       Xv = (uint4*)X;                      // 32 x uint4 per 256-f16 row
  Xv[e * 32 + q] = Hv[node * 16 + (q & 15)];
}

// ---------------------------------------------------------------- residual + col stats (P = H + P)
__global__ void k_resid_reduce(const float* __restrict__ H, float* __restrict__ P,
                               float* __restrict__ stats, int chunk) {
  int c  = threadIdx.x;                 // 128
  int r0 = blockIdx.x * chunk;
  int r1 = min(NP, r0 + chunk);
  float s = 0.f, s2 = 0.f;
  for (int r = r0; r < r1; ++r) {
    float v = H[r * DD + c] + P[r * DD + c];
    P[r * DD + c] = v;
    if (r < N_NODES) { s += v; s2 += v * v; }
  }
  atomicAdd(&stats[c], s);
  atomicAdd(&stats[128 + c], s2);
}

// ---------------------------------------------------------------- WMMA GEMM
// C[M,N] = A[M,K] @ Bt[N,K]^T + bias. One wave computes a 16x64 slab: FOUR
// independent accumulator chains so dependent WMMAs are 3 matrix-ops apart
// (covers the 1-NOP + 4-coexec f16 WMMA->WMMA RAW hazard with useful work),
// and the A fragment is loaded once per 32-K slab for 4 tiles.
// blockDim = (N/64)*32, gridDim.x = M/16, M multiple of 16.
// All strides compile-time so the epilogue is base + immediate-offset stores.
// MODE: 0 = f32 store; 1 = f32+f16 store; 2 = f32+2x f16 store;
//       3 = scatter-atomic rows into C via scat[] (fused segment_sum).
template <int K, int LDA, int LDC, int LDC16, int MODE, bool RELU, bool STATS>
__global__ void k_wmma_gemm(const _Float16* __restrict__ A,
                            const _Float16* __restrict__ Bt,
                            const float* __restrict__ bias,
                            float* __restrict__ C,
                            _Float16* __restrict__ C16,
                            _Float16* __restrict__ C16b,
                            const long long* __restrict__ scat,
                            float* __restrict__ stats, int rowsStat) {
  constexpr int NT = 4;                     // N-tiles per wave
  int wave = threadIdx.x >> 5;
  int lane = threadIdx.x & 31;
  int half = lane >> 4, lsub = lane & 15;
  int r0 = blockIdx.x * 16;
  int n0 = wave * (16 * NT);

  // 16-bit A/B fragment layout: lanes0-15 hold K 0-7 (v0-3) & 16-23 (v4-7);
  // lanes16-31 hold K 8-15 & 24-31 -> two contiguous 16B loads per 32-K slab.
  const _Float16* aptr = A + (r0 + lsub) * LDA + half * 8;
  const _Float16* bp[NT];
#pragma unroll
  for (int t = 0; t < NT; ++t) bp[t] = Bt + (n0 + t * 16 + lsub) * K + half * 8;

  v8f c[NT];
#pragma unroll
  for (int t = 0; t < NT; ++t) {
    float bv = bias[n0 + t * 16 + lsub];
#pragma unroll
    for (int j = 0; j < 8; ++j) c[t][j] = bv;
  }

#pragma unroll
  for (int k = 0; k < K; k += 32) {
    v16h a, b[NT];
    *(uint4*)&a         = *(const uint4*)(aptr + k);
    *(((uint4*)&a) + 1) = *(const uint4*)(aptr + k + 16);
#pragma unroll
    for (int t = 0; t < NT; ++t) {
      *(uint4*)&b[t]         = *(const uint4*)(bp[t] + k);
      *(((uint4*)&b[t]) + 1) = *(const uint4*)(bp[t] + k + 16);
    }
#pragma unroll
    for (int t = 0; t < NT; ++t)
      c[t] = __builtin_amdgcn_wmma_f32_16x16x32_f16(false, a, false, b[t], (short)0, c[t],
                                                    false, false);
  }

  int col0  = n0 + lsub;
  int rbase = r0 + half * 8;                // C layout: vgpr j = row rbase+j, col lsub

  if (MODE == 3) {
    // fused scatter-add (segment_sum over dst); vector-load the 8 dst indices
    int didx[8];
#pragma unroll
    for (int jj = 0; jj < 4; ++jj) {
      uint4 t = *(const uint4*)(scat + rbase + jj * 2);   // two little-endian int64s
      didx[jj * 2]     = (int)t.x;
      didx[jj * 2 + 1] = (int)t.z;
    }
#pragma unroll
    for (int j = 0; j < 8; ++j) {
      float* rowp = C + didx[j] * LDC + col0;
#pragma unroll
      for (int t = 0; t < NT; ++t) {
        float v = c[t][j];
        if (RELU) v = fmaxf(v, 0.f);
        atomicAdd(rowp + t * 16, v);
      }
    }
    return;
  }

  float*    cp   = C + rbase * LDC + col0;
  _Float16* cp16 = (MODE >= 1) ? (C16 + rbase * LDC16 + col0) : nullptr;
  _Float16* cpb  = (MODE == 2) ? (C16b + rbase * LDC16 + col0) : nullptr;
  float s[NT], q[NT];
#pragma unroll
  for (int t = 0; t < NT; ++t) { s[t] = 0.f; q[t] = 0.f; }
#pragma unroll
  for (int j = 0; j < 8; ++j) {
    bool live = (rbase + j) < rowsStat;
#pragma unroll
    for (int t = 0; t < NT; ++t) {
      float v = c[t][j];
      if (RELU) v = fmaxf(v, 0.f);
      cp[j * LDC + t * 16] = v;
      if (MODE >= 1) cp16[j * LDC16 + t * 16] = (_Float16)v;
      if (MODE == 2) cpb[j * LDC16 + t * 16]  = (_Float16)v;
      if (STATS && live) { s[t] += v; q[t] += v * v; }
    }
  }
  if (STATS) {                              // fused BatchNorm column statistics
#pragma unroll
    for (int t = 0; t < NT; ++t) {
      atomicAdd(&stats[col0 + t * 16], s[t]);
      atomicAdd(&stats[128 + col0 + t * 16], q[t]);
    }
  }
}

// ---------------------------------------------------------------- decoder L2 (64 -> 2) + log_softmax
__global__ void k_dec_out(const _Float16* __restrict__ D1n, const float* __restrict__ W,
                          const float* __restrict__ b, float* __restrict__ out) {
  int r = blockIdx.x * blockDim.x + threadIdx.x;
  if (r >= N_NODES) return;
  float z0 = b[0], z1 = b[1];
#pragma unroll
  for (int j = 0; j < HID; ++j) {
    float v = (float)D1n[r * HID + j];
    z0 += v * W[j * 2 + 0];
    z1 += v * W[j * 2 + 1];
  }
  float m = fmaxf(z0, z1);
  float l = m + logf(expf(z0 - m) + expf(z1 - m));
  out[r * 2 + 0] = z0 - l;
  out[r * 2 + 1] = z1 - l;
}

// ---------------------------------------------------------------- launcher
extern "C" void kernel_launch(void* const* d_in, const int* in_sizes, int n_in,
                              void* d_out, int out_size, void* d_ws, size_t ws_size,
                              hipStream_t stream) {
  // setup_inputs order: x, edge_index, num_nodes, params{enc,mem,msg,upd,norm,dec}
  const float*     x  = (const float*)d_in[0];
  const long long* ei = (const long long*)d_in[1];   // int64 [2,E]
  const float *encW1 = (const float*)d_in[3],  *encB1 = (const float*)d_in[4],
              *encG  = (const float*)d_in[5],  *encBe = (const float*)d_in[6],
              *encW2 = (const float*)d_in[7],  *encB2 = (const float*)d_in[8];
  const float *memW1 = (const float*)d_in[9],  *memB1 = (const float*)d_in[10],
              *memG  = (const float*)d_in[11], *memBe = (const float*)d_in[12],
              *memW2 = (const float*)d_in[13], *memB2 = (const float*)d_in[14];
  const float *msgW1 = (const float*)d_in[15], *msgB1 = (const float*)d_in[16],
              *msgG  = (const float*)d_in[17], *msgBe = (const float*)d_in[18],
              *msgW2 = (const float*)d_in[19], *msgB2 = (const float*)d_in[20];
  const float *updW1 = (const float*)d_in[21], *updB1 = (const float*)d_in[22],
              *updG  = (const float*)d_in[23], *updBe = (const float*)d_in[24],
              *updW2 = (const float*)d_in[25], *updB2 = (const float*)d_in[26];
  const float *normG = (const float*)d_in[27], *normBe = (const float*)d_in[28];
  const float *decW1 = (const float*)d_in[29], *decB1 = (const float*)d_in[30],
              *decG  = (const float*)d_in[31], *decBe = (const float*)d_in[32],
              *decW2 = (const float*)d_in[33], *decB2 = (const float*)d_in[34];

  // workspace bump allocator (all sizes multiples of 256 B)
  char* p = (char*)d_ws;
  auto bump = [&](size_t bytes) { char* q = p; p += bytes; return (void*)q; };
  _Float16* wt_mem1 = (_Float16*)bump(128 * 256 * 2);
  _Float16* wt_mem2 = (_Float16*)bump(128 * 128 * 2);
  _Float16* wt_msg1 = (_Float16*)bump(128 * 256 * 2);
  _Float16* wt_msg2 = (_Float16*)bump(128 * 128 * 2);
  _Float16* wt_upd1 = (_Float16*)bump(128 * 128 * 2);
  _Float16* wt_upd2 = (_Float16*)bump(128 * 128 * 2);
  _Float16* wt_enc2 = (_Float16*)bump(128 * 64 * 2);
  _Float16* wt_dec1 = (_Float16*)bump(64 * 128 * 2);
  _Float16* Xmem16  = (_Float16*)bump((size_t)NP * 256 * 2);       // [enc16 | h16]
  _Float16* Xmsg16  = (_Float16*)bump((size_t)N_EDGES * 256 * 2);  // [h[dst] | h[src]]
  float*    T1      = (float*)bump((size_t)N_EDGES * 128 * 4);     // layer-1 pre-BN
  _Float16* T1n16   = (_Float16*)bump((size_t)N_EDGES * 128 * 2);  // normalized f16
  float*    Hf32    = (float*)bump((size_t)NP * 128 * 4);          // h after mem-MLP
  _Float16* H16     = (_Float16*)bump((size_t)NP * 128 * 2);
  float*    Agg     = (float*)bump((size_t)NP * 128 * 4);
  _Float16* Agg16   = (_Float16*)bump((size_t)NP * 128 * 2);
  float*    U2      = (float*)bump((size_t)NP * 128 * 4);
  float*    stats   = (float*)bump(512 * 4);
  if ((size_t)(p - (char*)d_ws) > ws_size) return;   // insufficient scratch

  const int GN = NP / 16;        // 157 node tiles
  const int GE = N_EDGES / 16;   // 625 edge tiles
  const long long* noscat = nullptr;

  // -------- one-time setup (per launch; deterministic)
  k_zero_f32<<<2, 256, 0, stream>>>(stats, 512);
  k_zero_f32<<<(NP * 128 + 255) / 256, 256, 0, stream>>>(Agg, NP * 128);
  auto tr = [&](const float* W, _Float16* Bt, int din, int dout) {
    int n = din * dout;
    k_transpose_f16<<<(n + 255) / 256, 256, 0, stream>>>(W, Bt, din, dout);
  };
  tr(memW1, wt_mem1, 256, 128); tr(memW2, wt_mem2, 128, 128);
  tr(msgW1, wt_msg1, 256, 128); tr(msgW2, wt_msg2, 128, 128);
  tr(updW1, wt_upd1, 128, 128); tr(updW2, wt_upd2, 128, 128);
  tr(encW2, wt_enc2, 64, 128);  tr(decW1, wt_dec1, 128, 64);

  // -------- encoder: 2 -> 64 -> BN -> 128, ReLU; h0 = enc
  k_enc_l1<<<NP, 64, 0, stream>>>(x, encW1, encB1, T1);                   // T1 ld 64
  k_col_reduce<<<40, 64, 0, stream>>>(T1, 64, N_NODES, stats, 64);
  k_bn_finalize<<<1, 64, 0, stream>>>(stats, encG, encBe, (float)N_NODES);
  k_bn_apply_f16<64><<<(NP * 64 + 255) / 256, 256, 0, stream>>>(T1, 64, T1n16, 64, stats, NP);
  k_wmma_gemm<64, 64, 128, 256, 2, true, false><<<GN, 64, 0, stream>>>(
      T1n16, wt_enc2, encB2, Hf32, Xmem16, Xmem16 + 128, noscat, nullptr, 0);

  // -------- 150 message-passing iterations
  for (int it = 0; it < ITERS; ++it) {
    // h = mem_mlp(cat(enc, h))  : K=256 (stats fused) then K=128 (h16 mirror)
    k_wmma_gemm<256, 256, 128, 1, 0, true, true><<<GN, 64, 0, stream>>>(
        Xmem16, wt_mem1, memB1, T1, nullptr, nullptr, noscat, stats, N_NODES);
    k_bn_finalize<<<1, 128, 0, stream>>>(stats, memG, memBe, (float)N_NODES);
    k_bn_apply_f16<128><<<(NP * 128 + 255) / 256, 256, 0, stream>>>(T1, 128, T1n16, 128, stats, NP);
    k_wmma_gemm<128, 128, 128, 128, 1, true, false><<<GN, 64, 0, stream>>>(
        T1n16, wt_mem2, memB2, Hf32, H16, nullptr, noscat, nullptr, 0);
    // per-edge messages + fused segment_sum
    k_gather<<<(N_EDGES * 32) / 256, 256, 0, stream>>>(H16, ei, Xmsg16);
    k_wmma_gemm<256, 256, 128, 1, 0, true, true><<<GE, 64, 0, stream>>>(
        Xmsg16, wt_msg1, msgB1, T1, nullptr, nullptr, noscat, stats, N_EDGES);
    k_bn_finalize<<<1, 128, 0, stream>>>(stats, msgG, msgBe, (float)N_EDGES);
    k_bn_apply_f16<128><<<(N_EDGES * 128 + 255) / 256, 256, 0, stream>>>(T1, 128, T1n16, 128, stats, N_EDGES);
    k_wmma_gemm<128, 128, 128, 1, 3, false, false><<<GE, 64, 0, stream>>>(
        T1n16, wt_msg2, msgB2, Agg, nullptr, nullptr, ei + N_EDGES, nullptr, 0);
    // update MLP on aggregated messages (convert + re-zero Agg for next iter)
    k_f32_to_f16_zero<<<(NP * 128 + 255) / 256, 256, 0, stream>>>(Agg, Agg16, NP * 128);
    k_wmma_gemm<128, 128, 128, 1, 0, true, true><<<GN, 64, 0, stream>>>(
        Agg16, wt_upd1, updB1, T1, nullptr, nullptr, noscat, stats, N_NODES);
    k_bn_finalize<<<1, 128, 0, stream>>>(stats, updG, updBe, (float)N_NODES);
    k_bn_apply_f16<128><<<(NP * 128 + 255) / 256, 256, 0, stream>>>(T1, 128, T1n16, 128, stats, NP);
    k_wmma_gemm<128, 128, 128, 1, 0, false, false><<<GN, 64, 0, stream>>>(
        T1n16, wt_upd2, updB2, U2, nullptr, nullptr, noscat, nullptr, 0);
    // h = BN(h + upd(agg)); write h16 into Xmem right half for next iteration
    k_resid_reduce<<<40, 128, 0, stream>>>(Hf32, U2, stats, 64);
    k_bn_finalize<<<1, 128, 0, stream>>>(stats, normG, normBe, (float)N_NODES);
    k_bn_apply_f16<128><<<(NP * 128 + 255) / 256, 256, 0, stream>>>(U2, 128, Xmem16 + 128, 256, stats, NP);
  }

  // -------- decoder: 128 -> 64 -> BN -> 2 -> log_softmax
  k_wmma_gemm<128, 256, 64, 1, 0, true, true><<<GN, 32, 0, stream>>>(
      Xmem16 + 128, wt_dec1, decB1, T1, nullptr, nullptr, noscat, stats, N_NODES);
  k_bn_finalize<<<1, 64, 0, stream>>>(stats, decG, decBe, (float)N_NODES);
  k_bn_apply_f16<64><<<(NP * 64 + 255) / 256, 256, 0, stream>>>(T1, 64, T1n16, 64, stats, NP);
  k_dec_out<<<(N_NODES + 255) / 256, 256, 0, stream>>>(T1n16, decW2, decB2, (float*)d_out);
}
